// Base_OC_Module_61890478735834
// MI455X (gfx1250) — compile-verified, matrix operand order checked by reading, not run on hardware
//
#include <hip/hip_runtime.h>
#include <hip/hip_bf16.h>

// ---------------------------------------------------------------------------
// SAGAN self-attention block for MI455X (gfx1250), bf16 WMMA + flash softmax.
//   B=4, H=W=64 -> N=4096, C=512, dqk=64, dv=256.
// Pipeline:
//   prep:  x->bf16; pack [wf|wg|wh]^T; fold (wo,wc,BN,gamma) into Wa',Wx',b''
//   proj:  [16384x512]@[512x384] WMMA -> f,g (row major), hvT (transposed)
//   attn:  per-wave 16 queries x 128 dims, streaming softmax over 4096 keys,
//          S^T tiles via WMMA; C-layout of S^T == B-fragment layout for PV.
//   out:   y = relu(o@Wa' + x@Wx' + b'')  (concat+conv+BN+relu fused)
// GEMM K-loops are fully unrolled ping-pong pipelines (no loop-carried phis,
// so no rotation copies); sched_group_barrier directives pin the
// [4 WMMA][10 VMEM] stage interleave so each WMMA group runs with the next
// stages' loads still in flight instead of just-in-time wait-0 loading.
// ---------------------------------------------------------------------------

typedef __bf16 bf16;
typedef __attribute__((ext_vector_type(16))) __bf16 v16bf;
typedef __attribute__((ext_vector_type(8)))  __bf16 v8bf;
typedef __attribute__((ext_vector_type(8)))  float  v8f;

#define BN_EPS 1e-3f

#if __has_builtin(__builtin_amdgcn_sched_group_barrier)
#define SGB(mask, cnt) __builtin_amdgcn_sched_group_barrier((mask), (cnt), 0)
#else
#define SGB(mask, cnt)
#endif
// group masks: 0x008 = MFMA/WMMA, 0x020 = VMEM read

// Load one 16x32-bf16 A/B fragment row: lane half h (h8 = 8*h) picks
// K = kb + 8h + [0..7] (elems 0..7) and K = kb + 16 + 8h + [0..7] (elems 8..15).
__device__ __forceinline__ v16bf load_frag(const bf16* p, int h8) {
  v8bf lo = *(const v8bf*)(p + h8);
  v8bf hi = *(const v8bf*)(p + h8 + 16);
  return __builtin_shufflevector(lo, hi, 0, 1, 2, 3, 4, 5, 6, 7,
                                         8, 9, 10, 11, 12, 13, 14, 15);
}

__device__ __forceinline__ v8f wmma_bf16(v16bf a, v16bf b, v8f c) {
  // (neg_a, A, neg_b, B, c_mod, C, reuse_a, reuse_b)
  return __builtin_amdgcn_wmma_f32_16x16x32_bf16(false, a, false, b,
                                                 (short)0, c, false, false);
}

// ---------------------------- prep kernels ---------------------------------

__global__ void prep_xb(const float* __restrict__ x, bf16* __restrict__ xb, int n) {
  int i = blockIdx.x * blockDim.x + threadIdx.x;
  if (i < n) xb[i] = (bf16)x[i];
}

// wt[384][512] = [wf|wg|wh]^T (Cout-major rows, Cin contiguous), bias packed.
__global__ void prep_wt(const float* __restrict__ wf, const float* __restrict__ bf_,
                        const float* __restrict__ wg, const float* __restrict__ bg_,
                        const float* __restrict__ wh, const float* __restrict__ bh_,
                        bf16* __restrict__ wt, float* __restrict__ bias) {
  int idx = blockIdx.x * blockDim.x + threadIdx.x;
  if (idx >= 384 * 512) return;
  int j = idx >> 9, k = idx & 511;
  float v;
  if (j < 64)       v = wf[k * 64 + j];
  else if (j < 128) v = wg[k * 64 + (j - 64)];
  else              v = wh[k * 256 + (j - 128)];
  wt[idx] = (bf16)v;
  if (k == 0) bias[j] = (j < 64) ? bf_[j] : (j < 128) ? bg_[j - 64] : bh_[j - 128];
}

// WaT[512][256] = (gamma * s'[j]) * (wo @ wc_top)[d][j], transposed to j-major.
__global__ void prep_wa(const float* __restrict__ wo, const float* __restrict__ wc,
                        const float* __restrict__ gamma,
                        const float* __restrict__ bns, const float* __restrict__ bnv,
                        bf16* __restrict__ WaT) {
  int idx = blockIdx.x * blockDim.x + threadIdx.x;  // j*256 + d
  if (idx >= 512 * 256) return;
  int j = idx >> 8, d = idx & 255;
  float s = 0.f;
  for (int k = 0; k < 512; ++k) s += wo[d * 512 + k] * wc[k * 512 + j];
  float sp = bns[j] * rsqrtf(bnv[j] + BN_EPS);
  WaT[idx] = (bf16)(gamma[0] * sp * s);
}

// WxT[512][512] = s'[j] * (wc_top + wc_bot)[k][j], transposed to j-major.
__global__ void prep_wx(const float* __restrict__ wc,
                        const float* __restrict__ bns, const float* __restrict__ bnv,
                        bf16* __restrict__ WxT) {
  int idx = blockIdx.x * blockDim.x + threadIdx.x;  // j*512 + k
  if (idx >= 512 * 512) return;
  int j = idx >> 9, k = idx & 511;
  float sp = bns[j] * rsqrtf(bnv[j] + BN_EPS);
  WxT[idx] = (bf16)(sp * (wc[k * 512 + j] + wc[(512 + k) * 512 + j]));
}

// b''[j] = s'*(gamma*(bo@wc_top) + bc - mean) + bn_bias
__global__ void prep_b2(const float* __restrict__ bo, const float* __restrict__ wc,
                        const float* __restrict__ bc, const float* __restrict__ gamma,
                        const float* __restrict__ bns, const float* __restrict__ bnb,
                        const float* __restrict__ bnm, const float* __restrict__ bnv,
                        float* __restrict__ b2) {
  int j = blockIdx.x * blockDim.x + threadIdx.x;
  if (j >= 512) return;
  float s = 0.f;
  for (int k = 0; k < 512; ++k) s += bo[k] * wc[k * 512 + j];
  float sp = bns[j] * rsqrtf(bnv[j] + BN_EPS);
  b2[j] = sp * (gamma[0] * s + bc[j] - bnm[j]) + bnb[j];
}

// ------------------------- pipelined 16Mx64N K-block -----------------------
// Fully-unrolled ping-pong over K in steps of 64: consume set, reload set in
// place for K+64. All offsets are compile-time immediates; final reloads are
// statically elided. sched_group_barriers pin the stage interleave:
// [20 VMEM prologue] then per 32-K stage [4 WMMA][10 VMEM read] (the last two
// stages have no reload group).
template <int K, int BSTRIDE>
__device__ __forceinline__ void gemm_tile(const bf16* __restrict__ arow,
                                          const bf16* __restrict__ brow,
                                          int h8, v8f acc[4]) {
  v16bf aA = load_frag(arow, h8);
  v16bf aB = load_frag(arow + 32, h8);
  v16bf bA[4], bB[4];
#pragma unroll
  for (int t = 0; t < 4; ++t) bA[t] = load_frag(brow + t * (16 * BSTRIDE), h8);
#pragma unroll
  for (int t = 0; t < 4; ++t) bB[t] = load_frag(brow + t * (16 * BSTRIDE) + 32, h8);

#pragma unroll
  for (int kb = 0; kb < K; kb += 64) {
#pragma unroll
    for (int t = 0; t < 4; ++t) acc[t] = wmma_bf16(aA, bA[t], acc[t]);
    if (kb + 64 < K) {                   // reload A-set in place (folds away)
      aA = load_frag(arow + kb + 64, h8);
#pragma unroll
      for (int t = 0; t < 4; ++t)
        bA[t] = load_frag(brow + t * (16 * BSTRIDE) + kb + 64, h8);
    }
#pragma unroll
    for (int t = 0; t < 4; ++t) acc[t] = wmma_bf16(aB, bB[t], acc[t]);
    if (kb + 96 < K) {                   // reload B-set in place
      aB = load_frag(arow + kb + 96, h8);
#pragma unroll
      for (int t = 0; t < 4; ++t)
        bB[t] = load_frag(brow + t * (16 * BSTRIDE) + kb + 96, h8);
    }
  }

  // ---- scheduling directives: pin the software pipeline shape ----
  SGB(0x020, 20);                        // prologue loads
#pragma unroll
  for (int s = 0; s < K / 32; ++s) {
    SGB(0x008, 4);                       // stage s: 4 WMMAs
    if (s < K / 32 - 2) SGB(0x020, 10);  // stage s: reload for stage s+2
  }
}

// ------------------------- projection GEMM ---------------------------------
// [16384 x 512] @ [512 x 384] ; wave = 16M x 64N tile.
// cout<64 -> f rows; <128 -> g rows; >=128 -> hvT[d][n] (transposed store).
__global__ __launch_bounds__(256)
void proj_kernel(const bf16* __restrict__ xb, const bf16* __restrict__ wt,
                 const float* __restrict__ bias,
                 bf16* __restrict__ fq, bf16* __restrict__ gq, bf16* __restrict__ hvT) {
  const int lane = threadIdx.x & 31;
  const int r = lane & 15, h8 = (lane >> 4) << 3;
  const int w = blockIdx.x * (blockDim.x >> 5) + (threadIdx.x >> 5);
  const int m0 = (w / 6) * 16, n0 = (w % 6) * 64;
  v8f acc[4] = {};
  gemm_tile<512, 512>(xb + (m0 + r) * 512, wt + (n0 + r) * 512, h8, acc);

#pragma unroll
  for (int t = 0; t < 4; ++t) {
    int n = n0 + 16 * t + r;           // output column (per lane)
    float bv = bias[n];
#pragma unroll
    for (int v = 0; v < 8; ++v) {
      int m = m0 + h8 + v;             // output row (C layout)
      bf16 val = (bf16)(acc[t][v] + bv);
      if (n < 64)       fq[m * 64 + n] = val;
      else if (n < 128) gq[m * 64 + (n - 64)] = val;
      else {
        int batch = m >> 12, nr = m & 4095;
        hvT[((batch << 8) + (n - 128)) * 4096 + nr] = val;
      }
    }
  }
}

// --------------------------- flash attention -------------------------------
// Wave owns 16 queries x 128 value-dims. S^T = f @ g^T tiles; per 32-key
// chunk: online softmax (one shfl_xor(16) per reduction), probs repacked
// lane-locally into the B fragment of O^T += hvT @ P^T. f-fragments for the
// next chunk are reloaded in place right after their WMMAs and prefetch under
// the softmax + PV phases; unroll-by-2 lets the register generations
// alternate without rotation copies. hvT A-fragments are double-buffered
// inside the fully-unrolled PV loop.
__global__ __launch_bounds__(256)
void attn_kernel(const bf16* __restrict__ fq, const bf16* __restrict__ gq,
                 const bf16* __restrict__ hvT, bf16* __restrict__ o) {
  const int lane = threadIdx.x & 31;
  const int r = lane & 15, h8 = (lane >> 4) << 3;
  const int w = blockIdx.x * (blockDim.x >> 5) + (threadIdx.x >> 5);
  const int qblock = w >> 1, dh = w & 1;           // dim half: 0 or 1 (128 dims)
  const int batch = qblock >> 8, q0 = (qblock & 255) * 16;
  const int gb = batch << 12;

  const bf16* grow = gq + (gb + q0 + r) * 64;      // B operand: g rows = queries
  v16bf bg0 = load_frag(grow, h8);                 // head-dim K = 0..31
  v16bf bg1 = load_frag(grow + 32, h8);            // head-dim K = 32..63
  const bf16* fbase = fq + (gb + r) * 64;
  const bf16* hbase = hvT + ((batch << 8) + dh * 128) * 4096;

  v8f acc[8] = {};
  float mrow = -3.0e38f, lrow = 0.f;

  // prologue: fragments for key chunk 0 (keys r and r+16 of the chunk)
  v16bf af0a = load_frag(fbase, h8);
  v16bf af0b = load_frag(fbase + 32, h8);
  v16bf af1a = load_frag(fbase + 16 * 64, h8);
  v16bf af1b = load_frag(fbase + 16 * 64 + 32, h8);

#pragma unroll 2
  for (int kc = 0; kc < 4096; kc += 32) {
    v8f s0 = {};
    s0 = wmma_bf16(af0a, bg0, s0);
    s0 = wmma_bf16(af0b, bg1, s0);
    v8f s1 = {};
    s1 = wmma_bf16(af1a, bg0, s1);
    s1 = wmma_bf16(af1b, bg1, s1);

    // reload in place for the next chunk; overlaps softmax + PV below
    {
      const bf16* fn = fbase + ((kc + 32) & 4095) * 64;
      af0a = load_frag(fn, h8);
      af0b = load_frag(fn + 32, h8);
      af1a = load_frag(fn + 16 * 64, h8);
      af1b = load_frag(fn + 16 * 64 + 32, h8);
    }

    // per-query max over this lane's 16 scores + partner half-wave
    float mx = s0[0];
#pragma unroll
    for (int v = 1; v < 8; ++v) mx = fmaxf(mx, s0[v]);
#pragma unroll
    for (int v = 0; v < 8; ++v) mx = fmaxf(mx, s1[v]);
    mx = fmaxf(mx, __shfl_xor(mx, 16));
    float mnew = fmaxf(mrow, mx);
    float rescale = __expf(mrow - mnew);

    // exponentiate; pack lane-locally into B fragment for the PV WMMA:
    // elem v   <- key kc+8h+v      (S^T tile0, VGPR v)
    // elem v+8 <- key kc+16+8h+v   (S^T tile1, VGPR v)
    v16bf bp;
    float ls = 0.f;
#pragma unroll
    for (int v = 0; v < 8; ++v) {
      float p0 = __expf(s0[v] - mnew); ls += p0; bp[v] = (bf16)p0;
      float p1 = __expf(s1[v] - mnew); ls += p1; bp[v + 8] = (bf16)p1;
    }
    ls += __shfl_xor(ls, 16);
    lrow = lrow * rescale + ls;
    mrow = mnew;

#pragma unroll
    for (int t = 0; t < 8; ++t) {
#pragma unroll
      for (int v = 0; v < 8; ++v) acc[t][v] *= rescale;
    }

    // PV phase: double-buffered hvT A fragments (rows = value dims)
    v16bf av = load_frag(hbase + r * 4096 + kc, h8);
#pragma unroll
    for (int t = 0; t < 8; ++t) {
      v16bf avn;
      if (t < 7) avn = load_frag(hbase + ((t + 1) * 16 + r) * 4096 + kc, h8);
      acc[t] = wmma_bf16(av, bp, acc[t]);
      av = avn;
    }
  }

  float inv = 1.0f / lrow;
  bf16* orow = o + (gb + q0 + r) * 256 + dh * 128;  // O^T tile: lane = query col
#pragma unroll
  for (int t = 0; t < 8; ++t) {
#pragma unroll
    for (int v = 0; v < 8; ++v)
      orow[t * 16 + h8 + v] = (bf16)(acc[t][v] * inv);
  }
}

// ---------------------------- fused tail GEMM ------------------------------
// y = relu(o @ Wa' + x @ Wx' + b'')  [16384 x 512], K = 256 + 512.
__global__ __launch_bounds__(256)
void out_kernel(const bf16* __restrict__ ob, const bf16* __restrict__ xb,
                const bf16* __restrict__ WaT, const bf16* __restrict__ WxT,
                const float* __restrict__ b2, float* __restrict__ y) {
  const int lane = threadIdx.x & 31;
  const int r = lane & 15, h8 = (lane >> 4) << 3;
  const int w = blockIdx.x * (blockDim.x >> 5) + (threadIdx.x >> 5);
  const int m0 = (w >> 3) * 16, n0 = (w & 7) * 64;
  v8f acc[4] = {};
  gemm_tile<256, 256>(ob + (m0 + r) * 256, WaT + (n0 + r) * 256, h8, acc);
  gemm_tile<512, 512>(xb + (m0 + r) * 512, WxT + (n0 + r) * 512, h8, acc);

#pragma unroll
  for (int t = 0; t < 4; ++t) {
    int n = n0 + 16 * t + r;
    float bv = b2[n];
#pragma unroll
    for (int v = 0; v < 8; ++v)
      y[(m0 + h8 + v) * 512 + n] = fmaxf(acc[t][v] + bv, 0.f);
  }
}

// ------------------------------- launch ------------------------------------

extern "C" void kernel_launch(void* const* d_in, const int* in_sizes, int n_in,
                              void* d_out, int out_size, void* d_ws, size_t ws_size,
                              hipStream_t stream) {
  (void)in_sizes; (void)n_in; (void)out_size; (void)ws_size;
  const float* x     = (const float*)d_in[0];
  const float* wf    = (const float*)d_in[1];
  const float* bf_   = (const float*)d_in[2];
  const float* wg    = (const float*)d_in[3];
  const float* bg_   = (const float*)d_in[4];
  const float* wh    = (const float*)d_in[5];
  const float* bh_   = (const float*)d_in[6];
  const float* wo    = (const float*)d_in[7];
  const float* bo    = (const float*)d_in[8];
  const float* gamma = (const float*)d_in[9];
  const float* wc    = (const float*)d_in[10];
  const float* bc    = (const float*)d_in[11];
  const float* bns   = (const float*)d_in[12];
  const float* bnb   = (const float*)d_in[13];
  const float* bnm   = (const float*)d_in[14];
  const float* bnv   = (const float*)d_in[15];
  float* y = (float*)d_out;

  char* ws = (char*)d_ws;                       // ~39 MB total
  bf16*  XB   = (bf16*)(ws + 0);                // 16384*512 bf16   16.78 MB
  bf16*  F    = (bf16*)(ws + 16777216);         // 16384*64         2.10 MB
  bf16*  G    = (bf16*)(ws + 18874368);         // 16384*64         2.10 MB
  bf16*  HVT  = (bf16*)(ws + 20971520);         // 4*256*4096       8.39 MB
  bf16*  OB   = (bf16*)(ws + 29360128);         // 16384*256        8.39 MB
  bf16*  WT   = (bf16*)(ws + 37748736);         // 384*512          0.39 MB
  float* BIAS = (float*)(ws + 38141952);        // 384 f32
  bf16*  WAT  = (bf16*)(ws + 38143744);         // 512*256          0.26 MB
  bf16*  WXT  = (bf16*)(ws + 38405888);         // 512*512          0.52 MB
  float* B2   = (float*)(ws + 38930176);        // 512 f32

  prep_xb<<<32768, 256, 0, stream>>>(x, XB, 16384 * 512);
  prep_wt<<<768, 256, 0, stream>>>(wf, bf_, wg, bg_, wh, bh_, WT, BIAS);
  prep_wa<<<512, 256, 0, stream>>>(wo, wc, gamma, bns, bnv, WAT);
  prep_wx<<<1024, 256, 0, stream>>>(wc, bns, bnv, WXT);
  prep_b2<<<2, 256, 0, stream>>>(bo, wc, bc, gamma, bns, bnb, bnm, bnv, B2);

  proj_kernel<<<768, 256, 0, stream>>>(XB, WT, BIAS, F, G, HVT);   // 6144 waves
  attn_kernel<<<256, 256, 0, stream>>>(F, G, HVT, OB);             // 2048 waves
  out_kernel<<<1024, 256, 0, stream>>>(OB, XB, WAT, WXT, B2, y);   // 8192 waves
}
